// GraphBlock_59081570125129
// MI455X (gfx1250) — compile-verified, matrix operand order checked by reading, not run on hardware
//
#include <hip/hip_runtime.h>
#include <hip/hip_bf16.h>
#include <math.h>

// ---- problem constants (from reference) ----
#define NB    4096
#define VV    16
#define CC    512
#define HH    128
#define ROWS  (NB*VV)          // 65536 rows of h / hp
#define RB    64               // rows per block  -> 4 graphs per block
#define BB    4                // batches (graphs) per block
#define GRID1 (ROWS/RB)        // 1024 blocks
#define SXS   520              // LDS row stride (bf16) for 512-wide rows (pad 8, 16B aligned)
#define SQ    136              // LDS row stride (bf16) for 128-wide q/k rows
#define EPS_L1f 1e-12f
#define EPS_BNf 1e-5f
#define GAMMAf  1.0f
#define ALPHAf  1.0f

typedef __bf16 v16bf __attribute__((ext_vector_type(16)));
typedef __bf16 v8bf  __attribute__((ext_vector_type(8)));
typedef __bf16 v4bf  __attribute__((ext_vector_type(4)));
typedef float  v8f   __attribute__((ext_vector_type(8)));
typedef float  v4f   __attribute__((ext_vector_type(4)));

// Load one 16x32 bf16 fragment (A layout, or B supplied as NxK) per CDNA5 ISA 7.12.2:
// lane<16 holds row (lane), K = {0..7} and {16..23}; lane>=16 holds row(lane-16), K = {8..15} and {24..31}.
__device__ __forceinline__ v16bf load_frag(const __bf16* tile, int stride, int lane) {
    int r   = lane & 15;
    int sel = lane >> 4;
    const __bf16* p = tile + r * stride + sel * 8;
    v8bf lo = *(const v8bf*)(p);        // K = sel*8 .. sel*8+7
    v8bf hi = *(const v8bf*)(p + 16);   // K = 16+sel*8 .. 16+sel*8+7
    return __builtin_shufflevector(lo, hi, 0,1,2,3,4,5,6,7,8,9,10,11,12,13,14,15);
}

__device__ __forceinline__ v8f wmma_bf16(v16bf a, v16bf b, v8f c) {
    // D = A(16x32) * B(32x16) + C, f32 accumulate
    return __builtin_amdgcn_wmma_f32_16x16x32_bf16(false, a, false, b, (short)0, c, false, false);
}

// ---------------- fp32 -> bf16 conversion (weights) ----------------
__global__ void cvt_bf16_kernel(const float* __restrict__ s, __bf16* __restrict__ d, int n) {
    int i = blockIdx.x * blockDim.x + threadIdx.x;
    if (i < n) d[i] = (__bf16)s[i];
}

// ---------------- fused per-64-row kernel ----------------
__global__ __launch_bounds__(256, 1)
void fused_graphblock(const float* __restrict__ x, const float* __restrict__ adj,
                      const __bf16* __restrict__ Wb, const __bf16* __restrict__ Wqb,
                      const __bf16* __restrict__ Wkb, const float* __restrict__ bq,
                      const float* __restrict__ bk, float* __restrict__ hp,
                      float* __restrict__ psum, float* __restrict__ psq)
{
    __shared__ union U {
        __bf16 xs[RB * SXS];                                  // 66,560 B (phase 0/1)
        struct { __bf16 q[RB * SQ]; __bf16 k[RB * SQ]; } qk;  // 34,816 B (phase 2+) aliases xs
    } u;
    __shared__ __bf16 hs[RB * SXS];                           // 66,560 B  (h, bf16)
    __shared__ float  lg[BB][16][16];                         //  4,096 B  (logits -> graph)

    const int tid  = threadIdx.x;
    const int lane = tid & 31;
    const int wv   = tid >> 5;           // 8 wave32s
    const int blk  = blockIdx.x;

    // ---- phase 0: stage 64x512 fp32 rows of x into LDS as bf16 ----
    const float* xg = x + (size_t)blk * RB * CC;
    for (int i = tid; i < RB * CC / 4; i += 256) {
        v4f v = *(const v4f*)(xg + i * 4);
        int row = (i * 4) >> 9;          // /512
        int col = (i * 4) & (CC - 1);
        v4bf o = { (__bf16)v[0], (__bf16)v[1], (__bf16)v[2], (__bf16)v[3] };
        *(v4bf*)&u.xs[row * SXS + col] = o;
    }
    __syncthreads();

    // ---- phase 1: h = x * W^T  (64x512 = 4 Mtiles x 32 Ntiles, K=512) ----
    {
        int mt = wv >> 1, ng = wv & 1;   // wave -> (M-tile, half of N-tiles)
        const __bf16* arow = &u.xs[(mt * 16) * SXS];
        for (int nt = ng * 16; nt < ng * 16 + 16; ++nt) {
            v8f acc = {0.f,0.f,0.f,0.f,0.f,0.f,0.f,0.f};
            #pragma unroll
            for (int kk = 0; kk < 16; ++kk) {
                v16bf a = load_frag(arow + kk * 32, SXS, lane);
                v16bf b = load_frag(Wb + nt * 16 * CC + kk * 32, CC, lane); // W rows are output channels (NxK)
                acc = wmma_bf16(a, b, acc);
            }
            int sel = lane >> 4, col = nt * 16 + (lane & 15);
            #pragma unroll
            for (int j = 0; j < 8; ++j)                       // C layout: vgpr j -> rows j / j+8
                hs[(mt * 16 + j + 8 * sel) * SXS + col] = (__bf16)acc[j];
        }
    }
    __syncthreads();

    // ---- phase 2: q = h Wq^T + bq, k = h Wk^T + bk  (into LDS aliasing xs) ----
    {
        int mt = wv >> 1, isk = wv & 1;
        const __bf16* Wm  = isk ? Wkb : Wqb;
        const float* bias = isk ? bk : bq;
        __bf16* dst       = isk ? u.qk.k : u.qk.q;
        for (int nt = 0; nt < 8; ++nt) {
            v8f acc = {0.f,0.f,0.f,0.f,0.f,0.f,0.f,0.f};
            #pragma unroll
            for (int kk = 0; kk < 16; ++kk) {
                v16bf a = load_frag(&hs[(mt * 16) * SXS + kk * 32], SXS, lane);
                v16bf b = load_frag(Wm + nt * 16 * CC + kk * 32, CC, lane);
                acc = wmma_bf16(a, b, acc);
            }
            float bv = bias[nt * 16 + (lane & 15)];
            int sel = lane >> 4, col = nt * 16 + (lane & 15);
            #pragma unroll
            for (int j = 0; j < 8; ++j)
                dst[(mt * 16 + j + 8 * sel) * SQ + col] = (__bf16)(acc[j] + bv);
        }
    }
    __syncthreads();

    // ---- phase 3: logits[b] = q[b] * k[b]^T  (16x16, K=128 -> 4 WMMA), waves 0..3 ----
    if (wv < BB) {
        int b = wv;
        v8f acc = {0.f,0.f,0.f,0.f,0.f,0.f,0.f,0.f};
        #pragma unroll
        for (int kk = 0; kk < 4; ++kk) {
            v16bf a  = load_frag(&u.qk.q[(b * 16) * SQ + kk * 32], SQ, lane);
            v16bf bb = load_frag(&u.qk.k[(b * 16) * SQ + kk * 32], SQ, lane); // k rows as NxK
            acc = wmma_bf16(a, bb, acc);
        }
        int sel = lane >> 4;
        #pragma unroll
        for (int j = 0; j < 8; ++j)
            lg[b][j + 8 * sel][lane & 15] = acc[j];
    }
    __syncthreads();

    // ---- phase 4: row softmax + L1-normalized adj blend (one thread per (graph,row)) ----
    if (tid < BB * 16) {
        int b = tid >> 4, v = tid & 15;
        float lr[16], m = -INFINITY;
        #pragma unroll
        for (int w = 0; w < 16; ++w) { lr[w] = lg[b][v][w]; m = fmaxf(m, lr[w]); }
        float s = 0.f;
        #pragma unroll
        for (int w = 0; w < 16; ++w) { lr[w] = expf(lr[w] - m); s += lr[w]; }
        float invs = 1.f / s;
        const float* arow = adj + ((size_t)(blk * BB + b) * VV + v) * VV;
        float av[16], l1 = 0.f;
        #pragma unroll
        for (int w = 0; w < 16; ++w) { av[w] = arow[w]; l1 += fabsf(av[w]); }
        float il1 = 1.f / fmaxf(l1, EPS_L1f);
        #pragma unroll
        for (int w = 0; w < 16; ++w)
            lg[b][v][w] = (av[w] * il1 + GAMMAf * (lr[w] * invs)) * (1.f / (1.f + GAMMAf));
    }
    __syncthreads();

    // ---- phase 5: hp = graph @ h (K=16, VALU), ReLU, store + deterministic BN partials ----
    {
        int c0 = tid * 2;                               // 256 threads * 2 channels = 512
        float s0 = 0.f, s1 = 0.f, q0 = 0.f, q1 = 0.f;
        float* hpb = hp + (size_t)blk * RB * CC;
        for (int r = 0; r < RB; ++r) {
            int b = r >> 4, v = r & 15;
            const __bf16* hr = &hs[(b * 16) * SXS + c0];
            float a0 = 0.f, a1 = 0.f;
            #pragma unroll
            for (int w = 0; w < 16; ++w) {
                float g = lg[b][v][w];
                a0 = fmaf(g, (float)hr[w * SXS + 0], a0);
                a1 = fmaf(g, (float)hr[w * SXS + 1], a1);
            }
            a0 = fmaxf(a0, 0.f); a1 = fmaxf(a1, 0.f);   // ReLU before BN stats
            float2 st; st.x = a0; st.y = a1;
            *(float2*)&hpb[(size_t)r * CC + c0] = st;
            s0 += a0; s1 += a1; q0 += a0 * a0; q1 += a1 * a1;
        }
        // transposed partials: psum[c][blk] -> contiguous per-channel reduction later
        psum[(size_t)(c0    ) * GRID1 + blk] = s0;
        psum[(size_t)(c0 + 1) * GRID1 + blk] = s1;
        psq [(size_t)(c0    ) * GRID1 + blk] = q0;
        psq [(size_t)(c0 + 1) * GRID1 + blk] = q1;
    }
}

// ---------------- BN finalize: one block per channel, coalesced + deterministic tree ----------------
__global__ __launch_bounds__(256)
void bn_finalize(const float* __restrict__ psum, const float* __restrict__ psq,
                 const float* __restrict__ g, const float* __restrict__ b,
                 float* __restrict__ scale, float* __restrict__ shift)
{
    __shared__ float rs[256], rq[256];
    const int c   = blockIdx.x;           // 0..511
    const int tid = threadIdx.x;
    float s = 0.f, q = 0.f;
    #pragma unroll
    for (int t = tid; t < GRID1; t += 256) {   // 4 contiguous, coalesced reads each
        s += psum[(size_t)c * GRID1 + t];
        q += psq [(size_t)c * GRID1 + t];
    }
    rs[tid] = s; rq[tid] = q;
    __syncthreads();
    for (int off = 128; off > 0; off >>= 1) {
        if (tid < off) { rs[tid] += rs[tid + off]; rq[tid] += rq[tid + off]; }
        __syncthreads();
    }
    if (tid == 0) {
        const float invn = 1.f / (float)ROWS;
        float mean = rs[0] * invn;
        float var  = rq[0] * invn - mean * mean;   // biased var (matches reference)
        float inv  = rsqrtf(var + EPS_BNf);
        float sc   = inv * g[c];
        scale[c] = sc;
        shift[c] = b[c] - mean * sc;
    }
}

// ---------------- epilogue: out = x + ALPHA * (hp * scale + shift) ----------------
__global__ void epilogue(const float* __restrict__ x, const float* __restrict__ hp,
                         const float* __restrict__ sc, const float* __restrict__ sh,
                         float* __restrict__ out)
{
    size_t i = (size_t)blockIdx.x * 256 + threadIdx.x;   // float4 index
    int c0 = (int)((i * 4) & (CC - 1));
    v4f xv = *(const v4f*)(x  + i * 4);
    v4f hv = *(const v4f*)(hp + i * 4);
    v4f s  = *(const v4f*)(sc + c0);
    v4f t  = *(const v4f*)(sh + c0);
    v4f o  = xv + ALPHAf * (hv * s + t);
    *(v4f*)(out + i * 4) = o;
}

extern "C" void kernel_launch(void* const* d_in, const int* in_sizes, int n_in,
                              void* d_out, int out_size, void* d_ws, size_t ws_size,
                              hipStream_t stream) {
    const float* x    = (const float*)d_in[0];
    const float* adj  = (const float*)d_in[1];
    const float* W    = (const float*)d_in[2];
    const float* Wq   = (const float*)d_in[3];
    const float* bq   = (const float*)d_in[4];
    const float* Wk   = (const float*)d_in[5];
    const float* bk   = (const float*)d_in[6];
    const float* bn_g = (const float*)d_in[7];
    const float* bn_b = (const float*)d_in[8];
    (void)in_sizes; (void)n_in; (void)out_size; (void)ws_size;

    // workspace layout (bytes)
    char* w = (char*)d_ws;
    __bf16* Wb     = (__bf16*)(w + 0);                    // 524,288
    __bf16* Wqb    = (__bf16*)(w + 524288);               // 131,072
    __bf16* Wkb    = (__bf16*)(w + 655360);               // 131,072
    float*  psum   = (float*)(w + 786432);                // 512*1024*4 = 2,097,152 (transposed [c][blk])
    float*  psq    = (float*)(w + 786432 + 2097152);      // 2,097,152
    float*  bscale = (float*)(w + 4980736);               // 2,048
    float*  bshift = (float*)(w + 4982784);               // 2,048
    float*  hp     = (float*)(w + 5242880);               // 134,217,728 (pre-BN activations)

    cvt_bf16_kernel<<<(CC*CC + 255) / 256, 256, 0, stream>>>(W,  Wb,  CC*CC);
    cvt_bf16_kernel<<<(HH*CC + 255) / 256, 256, 0, stream>>>(Wq, Wqb, HH*CC);
    cvt_bf16_kernel<<<(HH*CC + 255) / 256, 256, 0, stream>>>(Wk, Wkb, HH*CC);

    fused_graphblock<<<GRID1, 256, 0, stream>>>(x, adj, Wb, Wqb, Wkb, bq, bk, hp, psum, psq);

    bn_finalize<<<CC, 256, 0, stream>>>(psum, psq, bn_g, bn_b, bscale, bshift);

    epilogue<<<(ROWS * CC / 4) / 256, 256, 0, stream>>>(x, hp, bscale, bshift, (float*)d_out);
}
